// ScaledDotProductAttention_84576495993326
// MI455X (gfx1250) — matrix operand order repaired and checked by reading, not compiled, not run on hardware
//
#include <hip/hip_runtime.h>
#include <math.h>

// ---------------------------------------------------------------------------
// Complex-QK attention for MI455X (gfx1250), single-pass design.
//   B=2 H=8 S=2048 D=64  (BH = 16 batch*head slices)
// One workgroup = (bh, 16 query rows). 256 threads = 8 wave32s.
// Dynamic LDS: 2 x [16 x 2048] f32 score rows = 256 KB (CDNA5 WGP has 320 KB).
// Phase 1: scores via v_wmma_f32_16x16x32_f16 (4 WMMA per 16x16 tile per
//          matrix, K=64), gate*|.| fused, written to LDS, row-max tracked.
// Phase 2: exact softmax (cross-wave max reduce, exp, ds_add_f32 row sums),
//          normalized attn streamed to HBM once, fully coalesced.
// Phase 3: PV GEMM from LDS scores (wave -> {real,imag} x {d-tile}), WMMA.
// HBM traffic ~= 537MB attn write + ~100MB reads -> ~28us floor @ 23.3 TB/s;
// f16 WMMA compute (57 GFLOP total) sits well under that floor.
// ---------------------------------------------------------------------------

typedef __attribute__((ext_vector_type(16))) _Float16 v16h;
typedef __attribute__((ext_vector_type(8)))  float    v8f;

#define S_LEN   2048
#define D_HEAD  64
#define N_BH    16
#define QTILE   16
#define NWAVES  8
#define COLS_PER_WAVE   (S_LEN / NWAVES)        /* 256 */
#define KTILES_PER_WAVE (COLS_PER_WAVE / 16)    /* 16  */
#define INV_TEMP 0.125f                          /* 1/sqrt(64) */

static __device__ __forceinline__ v8f wmma_f16(v16h a, v16h b, v8f c) {
  // (neg_a, A, neg_b, B, c_mod, C, reuse_a, reuse_b)
  return __builtin_amdgcn_wmma_f32_16x16x32_f16(false, a, false, b,
                                                (short)0, c, false, false);
}

__launch_bounds__(256, 1)
__global__ void cplx_attn_kernel(const float* __restrict__ qr,
                                 const float* __restrict__ qi,
                                 const float* __restrict__ kr,
                                 const float* __restrict__ ki,
                                 const float* __restrict__ vr,
                                 const float* __restrict__ vi,
                                 const float* __restrict__ dwm,
                                 const float* __restrict__ sigma,
                                 float* __restrict__ out_r,
                                 float* __restrict__ out_i,
                                 float* __restrict__ attn_r,
                                 float* __restrict__ attn_i) {
  extern __shared__ float lds[];              // [2][16][2048] scores (256 KB)
  __shared__ float redmx[2][NWAVES][16];      // per-wave row-max partials
  __shared__ float rowmax[2][16];
  __shared__ float rowsum[2][16];

  float* sc_r = lds;
  float* sc_i = lds + QTILE * S_LEN;

  const int wg   = blockIdx.x;                // bh*128 + qblock
  const int bh   = wg >> 7;
  const int qb   = wg & 127;
  const int tid  = threadIdx.x;
  const int wave = tid >> 5;
  const int lane = tid & 31;
  const int hl   = lane >> 4;                 // lane half (0|1)
  const int l16  = lane & 15;

  const size_t base = (size_t)bh * S_LEN * D_HEAD;
  const float inv_sig2 = 1.0f / (sigma[0] * sigma[0]);

  // ---- Phase 1a: Q A-fragments (16x32 f16; lane=row, K-halves split by hl) --
  // A layout: lanes 0-15 hold K = {k0+0..15}; lanes 16-31 hold
  //           K = {k0+16..31} for row m = lane%16.
  v16h ar0, ar1, ai0, ai1;
  {
    const float* qrp = qr + base + (size_t)(qb * QTILE + l16) * D_HEAD;
    const float* qip = qi + base + (size_t)(qb * QTILE + l16) * D_HEAD;
#pragma unroll
    for (int f = 0; f < 2; ++f) {
      const int k0 = f * 32 + hl * 16;
      v16h r, im;
#pragma unroll
      for (int j = 0; j < 8; ++j) {
        r[j]      = (_Float16)(qrp[k0 + j]      * INV_TEMP);
        r[8 + j]  = (_Float16)(qrp[k0 + 8 + j] * INV_TEMP);
        im[j]     = (_Float16)(qip[k0 + j]      * INV_TEMP);
        im[8 + j] = (_Float16)(qip[k0 + 8 + j] * INV_TEMP);
      }
      if (f == 0) { ar0 = r; ai0 = im; } else { ar1 = r; ai1 = im; }
    }
  }
  // f16 WMMA NEG bits are C-only (ISA 7.12), so pre-negate A for the -Qr*Ki term.
  const v16h arn0 = -ar0, arn1 = -ar1;

  // ---- Phase 1b: score tiles -> LDS, track per-row max --------------------
  float pmr[8], pmi[8];
#pragma unroll
  for (int v = 0; v < 8; ++v) { pmr[v] = -1e30f; pmi[v] = -1e30f; }

  const float* dwp = dwm + (size_t)(qb * QTILE) * S_LEN;

  for (int t = 0; t < KTILES_PER_WAVE; ++t) {
    const int kb = wave * COLS_PER_WAVE + t * 16;
    // B layout (32x16 f16): lane holds column c=l16 (key row kb+c),
    // K = f*32 + hl*16 + j, j=0..15 (contiguous in the key row).
    const float* krr = kr + base + (size_t)(kb + l16) * D_HEAD;
    const float* kri = ki + base + (size_t)(kb + l16) * D_HEAD;
    if (t + 1 < KTILES_PER_WAVE) {       // speculative L2 prefetch of next tile
      __builtin_prefetch(krr + D_HEAD * 16, 0, 0);
      __builtin_prefetch(kri + D_HEAD * 16, 0, 0);
    }
    v16h br0, br1, bi0, bi1;
#pragma unroll
    for (int j = 0; j < 16; ++j) {
      br0[j] = (_Float16)krr[hl * 16 + j];
      br1[j] = (_Float16)krr[32 + hl * 16 + j];
      bi0[j] = (_Float16)kri[hl * 16 + j];
      bi1[j] = (_Float16)kri[32 + hl * 16 + j];
    }
    v8f cr = {};
    cr = wmma_f16(ar0, br0, cr);
    cr = wmma_f16(ar1, br1, cr);
    cr = wmma_f16(ai0, bi0, cr);
    cr = wmma_f16(ai1, bi1, cr);
    v8f ci = {};
    ci = wmma_f16(ai0,  br0, ci);
    ci = wmma_f16(ai1,  br1, ci);
    ci = wmma_f16(arn0, bi0, ci);
    ci = wmma_f16(arn1, bi1, ci);

    // C/D layout: VGPR v, lanes 0-15 -> row v, lanes 16-31 -> row v+8; col=l16.
#pragma unroll
    for (int v = 0; v < 8; ++v) {
      const int row = v + hl * 8;
      const float g  = __expf(dwp[(size_t)row * S_LEN + kb + l16] * inv_sig2);
      const float xr = fabsf(cr[v] * g);
      const float xi = fabsf(ci[v] * g);
      sc_r[row * S_LEN + kb + l16] = xr;
      sc_i[row * S_LEN + kb + l16] = xi;
      pmr[v] = fmaxf(pmr[v], xr);
      pmi[v] = fmaxf(pmi[v], xi);
    }
  }

  // reduce row-max across the 16 lanes of each half (xor masks stay in-group)
#pragma unroll
  for (int m = 8; m >= 1; m >>= 1) {
#pragma unroll
    for (int v = 0; v < 8; ++v) {
      pmr[v] = fmaxf(pmr[v], __shfl_xor(pmr[v], m, 32));
      pmi[v] = fmaxf(pmi[v], __shfl_xor(pmi[v], m, 32));
    }
  }
  if (l16 == 0) {
#pragma unroll
    for (int v = 0; v < 8; ++v) {
      redmx[0][wave][hl * 8 + v] = pmr[v];
      redmx[1][wave][hl * 8 + v] = pmi[v];
    }
  }
  __syncthreads();

  if (tid < 32) {                 // 2 matrices x 16 rows
    const int mat = tid >> 4, row = tid & 15;
    float m = -1e30f;
#pragma unroll
    for (int w = 0; w < NWAVES; ++w) m = fmaxf(m, redmx[mat][w][row]);
    rowmax[mat][row] = m;
    rowsum[mat][row] = 0.0f;
  }
  __syncthreads();

  // ---- Phase 2: exp(x - max) in LDS + row sums (ds_add_f32) ---------------
  // Thread t owns columns {t, t+256, ..., t+1792} of every row: conflict-free
  // LDS access, one shared-atomic per wave per row.
  for (int row = 0; row < QTILE; ++row) {
    const float mr = rowmax[0][row], mi = rowmax[1][row];
    float sr = 0.0f, si = 0.0f;
#pragma unroll
    for (int j = 0; j < 8; ++j) {
      const int col = tid + 256 * j;
      const float er = __expf(sc_r[row * S_LEN + col] - mr);
      const float ei = __expf(sc_i[row * S_LEN + col] - mi);
      sc_r[row * S_LEN + col] = er;
      sc_i[row * S_LEN + col] = ei;
      sr += er; si += ei;
    }
#pragma unroll
    for (int m = 16; m >= 1; m >>= 1) {
      sr += __shfl_xor(sr, m, 32);
      si += __shfl_xor(si, m, 32);
    }
    if (lane == 0) {
      atomicAdd(&rowsum[0][row], sr);   // ds_add_f32
      atomicAdd(&rowsum[1][row], si);
    }
  }
  __syncthreads();

  // ---- Phase 2b: stream normalized attention to HBM (coalesced, once) ----
  for (int row = 0; row < QTILE; ++row) {
    const float rr = 1.0f / rowsum[0][row];
    const float ri = 1.0f / rowsum[1][row];
    const size_t g = ((size_t)bh * S_LEN + qb * QTILE + row) * S_LEN;
#pragma unroll
    for (int j = 0; j < 8; ++j) {
      const int col = tid + 256 * j;
      attn_r[g + col] = sc_r[row * S_LEN + col] * rr;
      attn_i[g + col] = sc_i[row * S_LEN + col] * ri;
    }
  }

  // ---- Phase 3: PV GEMM from LDS. wave -> (matrix, 16-wide d tile) --------
  {
    const int mat = wave >> 2;              // 0 = real, 1 = imag
    const int nt  = wave & 3;               // d-tile
    const float* sc   = mat ? sc_i : sc_r;
    const float* vsrc = mat ? vi : vr;
    float*       outp = mat ? out_i : out_r;

    v8f acc = {};
    for (int kk = 0; kk < S_LEN / 32; ++kk) {
      // A: 16x32 probs from LDS (row = l16), same A-fragment layout as above.
      const float* ap = sc + (size_t)l16 * S_LEN + kk * 32 + hl * 16;
      v16h a;
#pragma unroll
      for (int j = 0; j < 8; ++j) {
        a[j]     = (_Float16)ap[j];
        a[8 + j] = (_Float16)ap[8 + j];
      }
      // B: 32x16 slice of V; lane = output column d = nt*16 + l16,
      // K rows kk*32 + hl*16 + j (per-j loads are 64B coalesced across lanes).
      const float* bp = vsrc + base + (size_t)(kk * 32 + hl * 16) * D_HEAD
                        + nt * 16 + l16;
      v16h b;
#pragma unroll
      for (int j = 0; j < 16; ++j) b[j] = (_Float16)bp[(size_t)j * D_HEAD];
      acc = wmma_f16(a, b, acc);
    }
#pragma unroll
    for (int v = 0; v < 8; ++v) {
      const int row = v + hl * 8;
      const float o = acc[v] / rowsum[mat][row];
      outp[((size_t)bh * S_LEN + qb * QTILE + row) * D_HEAD + nt * 16 + l16] = o;
    }
  }
}

// ---------------------------------------------------------------------------
extern "C" void kernel_launch(void* const* d_in, const int* in_sizes, int n_in,
                              void* d_out, int out_size, void* d_ws, size_t ws_size,
                              hipStream_t stream) {
  const float* qr    = (const float*)d_in[0];
  const float* qi    = (const float*)d_in[1];
  const float* kr    = (const float*)d_in[2];
  const float* ki    = (const float*)d_in[3];
  const float* vr    = (const float*)d_in[4];
  const float* vi    = (const float*)d_in[5];
  const float* dwm   = (const float*)d_in[6];
  const float* sigma = (const float*)d_in[7];

  float* out = (float*)d_out;
  const size_t od = (size_t)N_BH * S_LEN * D_HEAD;   // 2,097,152
  const size_t oa = (size_t)N_BH * S_LEN * S_LEN;    // 67,108,864
  float* out_r  = out;
  float* out_i  = out + od;
  float* attn_r = out + 2 * od;
  float* attn_i = out + 2 * od + oa;

  // 256 KB dynamic LDS per workgroup (CDNA5 WGP: 320 KB available).
  const size_t shmem = 2u * QTILE * S_LEN * sizeof(float);
  dim3 grid(N_BH * (S_LEN / QTILE));   // 16 * 128 = 2048 workgroups
  cplx_attn_kernel<<<grid, 256, shmem, stream>>>(qr, qi, kr, ki, vr, vi, dwm,
                                                 sigma, out_r, out_i,
                                                 attn_r, attn_i);
}
